// GRU_90417651516547
// MI455X (gfx1250) — matrix operand order repaired and checked by reading, not compiled, not run on hardware
//
#include <hip/hip_runtime.h>
#include <hip/hip_bf16.h>
#include <math.h>

// ---------------------------------------------------------------------------
// Model dims (compile-time constants from the reference)
#define BB   256
#define SS   512
#define EE   128
#define HH   128
#define H3   384     // 3*H, gate order z|r|h
#define CC   4
#define BN_EPS 1e-3f

typedef __attribute__((ext_vector_type(16))) __bf16 v16bf;
typedef __attribute__((ext_vector_type(8)))  float  v8f;

union BFrag {
    v16bf bf;
    unsigned u[8];
    uint4 q[2];
};

// round-to-nearest-even f32 -> bf16 bits
__device__ __forceinline__ unsigned short f2bf(float f) {
    unsigned u = __float_as_uint(f);
    u += 0x7fffu + ((u >> 16) & 1u);
    return (unsigned short)(u >> 16);
}
__device__ __forceinline__ unsigned pack2bf(float lo, float hi) {
#if __has_builtin(__builtin_amdgcn_cvt_pk_bf16_f32)
    typedef __attribute__((ext_vector_type(2))) __bf16 v2bf;
    union { v2bf b; unsigned u; } cv;
    cv.b = __builtin_amdgcn_cvt_pk_bf16_f32(lo, hi);
    return cv.u;
#else
    return (unsigned)f2bf(lo) | ((unsigned)f2bf(hi) << 16);
#endif
}
__device__ __forceinline__ float bf2f(unsigned short b) {
    return __uint_as_float(((unsigned)b) << 16);
}
__device__ __forceinline__ float sigmoidf_(float x) {
    return 1.0f / (1.0f + __expf(-x));
}

__device__ __forceinline__ v8f wmma_bf16(const BFrag& a, const BFrag& b, v8f c) {
    return __builtin_amdgcn_wmma_f32_16x16x32_bf16(
        /*neg_a=*/false, a.bf, /*neg_b=*/false, b.bf,
        /*c_mod=*/(short)0, c, /*reuse_a=*/false, /*reuse_b=*/false);
}

// ---- gfx1250 async global->LDS copy (ASYNCcnt path) ------------------------
// LDS offset of a __shared__ object == low 32 bits of its flat address
// (see ISA 10.2, LDS aperture).
__device__ __forceinline__ unsigned lds_off(const void* p) {
    return (unsigned)(uintptr_t)p;
}
__device__ __forceinline__ void async_copy_b128(unsigned ldsAddr, const void* gbase,
                                                unsigned voff) {
    asm volatile("global_load_async_to_lds_b128 %0, %1, %2"
                 :
                 : "v"(ldsAddr), "v"(voff), "s"(gbase)
                 : "memory");
}
__device__ __forceinline__ void wait_asynccnt0() {
    asm volatile("s_wait_asynccnt 0" ::: "memory");
}

// Pack a [128 x 384] f32 row-major weight matrix into LDS as bf16 WMMA-B
// fragments: slot = ((ntile*4 + kchunk)*32 + lane)*8 + v, each slot holds
// bf16 pair (k, k+1) with k = kchunk*32 + (v>=4 ? 16 : 0) + (lane>=16 ? 8 : 0) + 2(v&3),
// n = ntile*16 + (lane & 15).
#define WSLOTS (24 * 4 * 32 * 8)   // 24576 u32 = 96 KB
__device__ __forceinline__ void pack_weights_to_lds(const float* __restrict__ W,
                                                    unsigned* lds, int tid) {
    for (int s = tid; s < WSLOTS; s += 256) {
        int v     = s & 7;
        int lane  = (s >> 3) & 31;
        int chunk = (s >> 8) & 3;
        int ntile = s >> 10;
        int n = ntile * 16 + (lane & 15);
        int k = chunk * 32 + ((v & 4) ? 16 : 0) + ((lane >> 4) << 3) + (v & 3) * 2;
        lds[s] = pack2bf(W[k * H3 + n], W[(k + 1) * H3 + n]);
    }
}

__device__ __forceinline__ BFrag load_B(const unsigned* lds, int ntile, int chunk, int lane) {
    BFrag f;
    const uint4* p = (const uint4*)&lds[(((ntile * 4 + chunk) * 32) + lane) * 8];
    f.q[0] = p[0];
    f.q[1] = p[1];
    return f;
}

// ---------------------------------------------------------------------------
// Phase 1: xp[t][b][0..383] = embed[x[b][t]] @ Wi + b[0]   (bf16, TIME-MAJOR
// so phase 2 reads one contiguous 12KB tile per block per step).
// Grid: 1024 blocks x 256 threads; wave handles one 16-row tile x 24 N-tiles.
__global__ __launch_bounds__(256) void k_xproj(const int* __restrict__ x,
                                               const float* __restrict__ embed,
                                               const float* __restrict__ Wi,
                                               const float* __restrict__ bias0,
                                               unsigned short* __restrict__ xp) {
    __shared__ unsigned ldsW[WSLOTS];        // 96 KB: Wi in B-fragment layout
    const int tid = threadIdx.x;
    pack_weights_to_lds(Wi, ldsW, tid);
    __syncthreads();

    const int wave = tid >> 5, lane = tid & 31;
    const int m = lane & 15, khalf = lane >> 4;
    const int row0 = (blockIdx.x * 8 + wave) * 16;   // flat row = b*512 + t
    const int bidx = row0 >> 9;                       // batch index (tiles never span b)
    const int t0   = row0 & 511;                      // first timestep of tile
    const int xrow = x[row0 + m];                     // embedding row for lane's M

    // A fragments for K = 0..127 (4 chunks of 32), bf16, per 05_wmma layout
    BFrag A[4];
    const float2* er = (const float2*)(embed + (size_t)xrow * EE);
#pragma unroll
    for (int c = 0; c < 4; ++c) {
#pragma unroll
        for (int v = 0; v < 8; ++v) {
            int k = 32 * c + ((v & 4) ? 16 : 0) + khalf * 8 + (v & 3) * 2;
            float2 e = er[k >> 1];
            A[c].u[v] = pack2bf(e.x, e.y);
        }
    }

#pragma unroll 1
    for (int j = 0; j < 24; ++j) {
        v8f acc = {};
#pragma unroll
        for (int c = 0; c < 4; ++c) {
            BFrag Bf = load_B(ldsW, j, c, lane);
            acc = wmma_bf16(A[c], Bf, acc);
        }
        const int n = j * 16 + m;       // C/D layout: lane -> column n
        const float bv = bias0[n];
#pragma unroll
        for (int v = 0; v < 8; ++v) {
            int mm = v + (khalf ? 8 : 0);             // row in tile => timestep t0+mm
            xp[((size_t)(t0 + mm) * BB + bidx) * H3 + n] = f2bf(acc[v] + bv);
        }
    }
}

// ---------------------------------------------------------------------------
// Phase 2: persistent GRU scan. Grid: 16 blocks x 256 threads; block owns a
// 16-row batch tile; wave w owns hidden columns [16w, 16w+16) => z-tile w,
// r-tile 8+w, h-tile 16+w. Wh B-fragments held in registers for all 512
// steps; per-step x projections double-buffered into LDS with async DMA.
__global__ __launch_bounds__(256) void k_gru(const unsigned short* __restrict__ xp,
                                             const float* __restrict__ Wh,
                                             const float* __restrict__ bias, // [2][384]
                                             float* __restrict__ hout) {
    __shared__ unsigned ldsW[WSLOTS];                       // 96 KB
    __shared__ __align__(16) unsigned short hbuf[16 * HH];  // 4 KB, h_t in bf16
    __shared__ __align__(16) unsigned short xbuf[2][16 * H3]; // 2 x 12 KB x tiles
    const int tid = threadIdx.x, wave = tid >> 5, lane = tid & 31;
    const int b0 = blockIdx.x * 16;

    pack_weights_to_lds(Wh, ldsW, tid);
    for (int s = tid; s < 16 * HH; s += 256) hbuf[s] = 0;  // h0 = 0
    __syncthreads();

    BFrag Bz[4], Br[4], Bh[4];
#pragma unroll
    for (int c = 0; c < 4; ++c) {
        Bz[c] = load_B(ldsW, wave, c, lane);
        Br[c] = load_B(ldsW, 8 + wave, c, lane);
        Bh[c] = load_B(ldsW, 16 + wave, c, lane);
    }

    const int m = lane & 15, khalf = lane >> 4;
    const int ncol = wave * 16 + m;              // owned hidden column
    const float bz = bias[H3 + ncol];
    const float br = bias[H3 + HH + ncol];
    const float bh = bias[H3 + 2 * HH + ncol];

    // issue async prefetch of step-0 x tile: 16*384 bf16 = 12288 B contiguous,
    // 3 x b128 per thread
    {
        const char* g0 = (const char*)(xp + ((size_t)0 * BB + b0) * H3);
        unsigned lb = lds_off(&xbuf[0][0]);
#pragma unroll
        for (int c = 0; c < 3; ++c) {
            unsigned off = (unsigned)(tid * 16 + c * 4096);
            async_copy_b128(lb + off, g0, off);
        }
    }

    v8f h = {};                                  // C/D-layout fragment of h

#pragma unroll 1
    for (int t = 0; t < SS; ++t) {
        const int cur = t & 1;
        wait_asynccnt0();
        __syncthreads();   // (1) x tile t + h_t fully visible to all waves

        if (t + 1 < SS) {  // prefetch next step's tile into the other buffer
            const char* g = (const char*)(xp + ((size_t)(t + 1) * BB + b0) * H3);
            unsigned lb = lds_off(&xbuf[cur ^ 1][0]);
#pragma unroll
            for (int c = 0; c < 3; ++c) {
                unsigned off = (unsigned)(tid * 16 + c * 4096);
                async_copy_b128(lb + off, g, off);
            }
        }

        // h_t -> A fragments (16-bit A layout: contiguous K pairs per lane)
        BFrag A[4];
#pragma unroll
        for (int c = 0; c < 4; ++c) {
            A[c].q[0] = *(const uint4*)&hbuf[m * HH + 32 * c + khalf * 8];
            A[c].q[1] = *(const uint4*)&hbuf[m * HH + 32 * c + 16 + khalf * 8];
        }
        // gate inputs for this lane's columns from the LDS x tile
        float xz[8], xr[8], xh[8];
#pragma unroll
        for (int v = 0; v < 8; ++v) {
            int mm = v + (khalf ? 8 : 0);
            xz[v] = bf2f(xbuf[cur][mm * H3 + ncol]);
            xr[v] = bf2f(xbuf[cur][mm * H3 + HH + ncol]);
            xh[v] = bf2f(xbuf[cur][mm * H3 + 2 * HH + ncol]);
        }
        __syncthreads();   // (2) all waves done reading hbuf; safe to overwrite

        v8f az = {}, ar = {}, ah = {};
#pragma unroll
        for (int c = 0; c < 4; ++c) az = wmma_bf16(A[c], Bz[c], az);
#pragma unroll
        for (int c = 0; c < 4; ++c) ar = wmma_bf16(A[c], Br[c], ar);
#pragma unroll
        for (int c = 0; c < 4; ++c) ah = wmma_bf16(A[c], Bh[c], ah);

#pragma unroll
        for (int v = 0; v < 8; ++v) {
            float z  = sigmoidf_(xz[v] + az[v] + bz);
            float r  = sigmoidf_(xr[v] + ar[v] + br);
            float hh = tanhf(xh[v] + r * (ah[v] + bh));
            h[v] = z * h[v] + (1.0f - z) * hh;
        }
        // publish h_{t+1}
#pragma unroll
        for (int v = 0; v < 8; ++v) {
            int mm = v + (khalf ? 8 : 0);
            hbuf[mm * HH + ncol] = f2bf(h[v]);
        }
        __syncthreads();   // h_{t+1} stores ordered before next iteration's reads
    }

#pragma unroll
    for (int v = 0; v < 8; ++v) {
        int mm = v + (khalf ? 8 : 0);
        hout[(size_t)(b0 + mm) * HH + ncol] = h[v];
    }
}

// ---------------------------------------------------------------------------
// Phase 3: BN1+ReLU -> FC1 -> BN2+ReLU -> FC2 -> softmax. One block per row.
__global__ __launch_bounds__(256) void k_tail(const float* __restrict__ hin,
                                              const float* __restrict__ fc1_w,
                                              const float* __restrict__ fc1_b,
                                              const float* __restrict__ fc2_w,
                                              const float* __restrict__ fc2_b,
                                              const float* __restrict__ g1,
                                              const float* __restrict__ be1,
                                              const float* __restrict__ m1,
                                              const float* __restrict__ v1,
                                              const float* __restrict__ g2,
                                              const float* __restrict__ be2,
                                              const float* __restrict__ m2,
                                              const float* __restrict__ v2,
                                              float* __restrict__ out) {
    __shared__ float sh[HH];
    __shared__ float sa[200];
    __shared__ float so[CC];
    const int row = blockIdx.x, tid = threadIdx.x;

    if (tid < HH) {
        float hv = hin[(size_t)row * HH + tid];
        hv = (hv - m1[tid]) * rsqrtf(v1[tid] + BN_EPS) * g1[tid] + be1[tid];
        sh[tid] = fmaxf(hv, 0.0f);
    }
    __syncthreads();
    if (tid < 200) {
        float acc = fc1_b[tid];
#pragma unroll 4
        for (int k = 0; k < HH; ++k) acc += sh[k] * fc1_w[k * 200 + tid];
        acc = (acc - m2[tid]) * rsqrtf(v2[tid] + BN_EPS) * g2[tid] + be2[tid];
        sa[tid] = fmaxf(acc, 0.0f);
    }
    __syncthreads();
    if (tid < CC) {
        float acc = fc2_b[tid];
#pragma unroll 4
        for (int j = 0; j < 200; ++j) acc += sa[j] * fc2_w[j * CC + tid];
        so[tid] = acc;
    }
    __syncthreads();
    if (tid == 0) {
        float mx = fmaxf(fmaxf(so[0], so[1]), fmaxf(so[2], so[3]));
        float e0 = __expf(so[0] - mx), e1 = __expf(so[1] - mx);
        float e2 = __expf(so[2] - mx), e3 = __expf(so[3] - mx);
        float s = e0 + e1 + e2 + e3;
        out[(size_t)row * CC + 0] = e0 / s;
        out[(size_t)row * CC + 1] = e1 / s;
        out[(size_t)row * CC + 2] = e2 / s;
        out[(size_t)row * CC + 3] = e3 / s;
    }
}

// ---------------------------------------------------------------------------
extern "C" void kernel_launch(void* const* d_in, const int* in_sizes, int n_in,
                              void* d_out, int out_size, void* d_ws, size_t ws_size,
                              hipStream_t stream) {
    const int*   x      = (const int*)  d_in[0];
    const float* embed  = (const float*)d_in[1];
    const float* Wi     = (const float*)d_in[2];
    const float* Wh     = (const float*)d_in[3];
    const float* b      = (const float*)d_in[4];   // [2][384]
    const float* fc1_w  = (const float*)d_in[5];
    const float* fc1_b  = (const float*)d_in[6];
    const float* fc2_w  = (const float*)d_in[7];
    const float* fc2_b  = (const float*)d_in[8];
    const float* bn1_g  = (const float*)d_in[9];
    const float* bn1_b  = (const float*)d_in[10];
    const float* bn1_m  = (const float*)d_in[11];
    const float* bn1_v  = (const float*)d_in[12];
    const float* bn2_g  = (const float*)d_in[13];
    const float* bn2_b  = (const float*)d_in[14];
    const float* bn2_m  = (const float*)d_in[15];
    const float* bn2_v  = (const float*)d_in[16];

    // workspace: xp (bf16, time-major [S][B][384]) then h_last (f32, [B][H])
    unsigned short* xp = (unsigned short*)d_ws;
    const size_t xp_bytes = (size_t)BB * SS * H3 * sizeof(unsigned short); // 96 MiB
    float* hws = (float*)((char*)d_ws + xp_bytes);

    k_xproj<<<1024, 256, 0, stream>>>(x, embed, Wi, b, xp);
    k_gru<<<16, 256, 0, stream>>>(xp, Wh, b, hws);
    k_tail<<<BB, 256, 0, stream>>>(hws, fc1_w, fc1_b, fc2_w, fc2_b,
                                   bn1_g, bn1_b, bn1_m, bn1_v,
                                   bn2_g, bn2_b, bn2_m, bn2_v,
                                   (float*)d_out);
}
